// SenderAgent_19980187861589
// MI455X (gfx1250) — matrix operand order repaired and checked by reading, not compiled, not run on hardware
//
#include <hip/hip_runtime.h>
#include <hip/hip_bf16.h>

// ---------------------------------------------------------------------------
// RGCN x2 -> mean pool -> dense heads -> log_softmax, for MI455X (gfx1250).
// Node GEMMs run on v_wmma_f32_16x16x32_bf16 (wave32, one 16x16 tile / wave).
// Edge aggregation = f32 global atomics (bandwidth bound, ~2.4 GB traffic).
// ---------------------------------------------------------------------------

typedef __bf16 bf16;
typedef __attribute__((ext_vector_type(16))) __bf16 v16bf;
typedef __attribute__((ext_vector_type(8)))  __bf16 bf16x8;
typedef __attribute__((ext_vector_type(8)))  float  v8f;

#define N_NODES 50000
#define N_EDGES 800000
#define N_GRAPHS 64
#define FEAT 128
#define EMB 256
#define HID 512
#define VOCAB 512

// ---------------- utility kernels ----------------

__global__ void k_zero_f32(float* __restrict__ p, long n) {
    long i = (long)blockIdx.x * blockDim.x + threadIdx.x;
    if (i < n) p[i] = 0.0f;
}

__global__ void k_cvt_bf16(const float* __restrict__ in, bf16* __restrict__ out, long n) {
    long i = (long)blockIdx.x * blockDim.x + threadIdx.x;
    if (i < n) out[i] = (bf16)in[i];
}

// Repack row-major f32 weight W[K,N] into the per-lane WMMA B-operand layout:
// for (tile_n, kstep s): 32 lanes x 16 halves contiguous (1 KB per K-step).
// Lane l holds column n = tile_n*16 + (l&15); half j holds K = s*32 + ((l>=16)?8:0) + (j&7) + (j>=8?16:0).
__global__ void k_pack_b(const float* __restrict__ W, bf16* __restrict__ Bp, int K, int N) {
    int idx = blockIdx.x * blockDim.x + threadIdx.x;
    if (idx >= K * N) return;
    int j    = idx & 15;
    int lane = (idx >> 4) & 31;
    int sl   = idx >> 9;
    int ksteps = K >> 5;
    int s  = sl % ksteps;
    int tn = sl / ksteps;
    int kk  = s * 32 + ((lane >> 4) << 3) + (j & 7) + ((j >> 3) << 4);
    int col = tn * 16 + (lane & 15);
    Bp[idx] = (bf16)W[kk * N + col];
}

// ---------------- WMMA GEMM: one wave per 16x16 output tile ----------------
// MODE 0: Y = A @ B                 (f32 store, feeds edge scatter)
// MODE 1: H = relu(A@B + agg/cnt + b)  -> bf16 store (layer-1 fused epilogue)
// MODE 2: same value, atomicAdd into pool_s[batch[row]]  (layer-2 fused epilogue)
template <int MODE>
__global__ void k_gemm_wmma(const bf16* __restrict__ A, const bf16* __restrict__ Bp,
                            float* __restrict__ Yout,
                            const float* __restrict__ agg, const float* __restrict__ cnt,
                            const float* __restrict__ bias,
                            bf16* __restrict__ Hout,
                            const int* __restrict__ batch, float* __restrict__ pool_s,
                            int M, int N, int K) {
    int wave = (int)((blockIdx.x * blockDim.x + threadIdx.x) >> 5);
    int lane = threadIdx.x & 31;
    int ntiles = N >> 4;
    int mtiles = M >> 4;
    if (wave >= ntiles * mtiles) return;          // whole-wave exit: EXEC all-1s inside
    int tm = wave / ntiles;
    int tn = wave % ntiles;
    int ksteps = K >> 5;

    // A operand: lane holds row (lane&15); halves j=0..7 -> K=koff..koff+7,
    // j=8..15 -> K=koff+16..koff+23, koff = (lane>=16 ? 8 : 0).
    int arow  = tm * 16 + (lane & 15);
    int koff0 = (lane >> 4) << 3;
    const bf16* aptr = A + (size_t)arow * K + koff0;
    const bf16* bptr = Bp + (((size_t)tn * ksteps) << 9) + ((size_t)lane << 4);

    v8f c = {};
    for (int s = 0; s < ksteps; ++s) {
        bf16x8 a0 = *(const bf16x8*)(aptr + s * 32);
        bf16x8 a1 = *(const bf16x8*)(aptr + s * 32 + 16);
        v16bf a, b;
        b = *(const v16bf*)(bptr + ((size_t)s << 9));
#pragma unroll
        for (int j = 0; j < 8; ++j) { a[j] = a0[j]; a[j + 8] = a1[j]; }
        c = __builtin_amdgcn_wmma_f32_16x16x32_bf16(false, a, false, b,
                                                    (short)0, c, false, false);
    }

    // C/D layout: VGPR i, lane l -> M = i + 8*(l>>4), N = l&15 (within tile).
    int col   = tn * 16 + (lane & 15);
    int rbase = tm * 16 + ((lane >> 4) << 3);
    if (MODE == 0) {
#pragma unroll
        for (int i = 0; i < 8; ++i)
            Yout[(size_t)(rbase + i) * N + col] = c[i];
    } else if (MODE == 1) {
        float bb = bias[col];
#pragma unroll
        for (int i = 0; i < 8; ++i) {
            int r = rbase + i;
            float v = c[i] + agg[(size_t)r * N + col] / fmaxf(cnt[r], 1.0f) + bb;
            Hout[(size_t)r * N + col] = (bf16)fmaxf(v, 0.0f);
        }
    } else {
        float bb = bias[col];
#pragma unroll
        for (int i = 0; i < 8; ++i) {
            int r = rbase + i;
            float v = c[i] + agg[(size_t)r * N + col] / fmaxf(cnt[r], 1.0f) + bb;
            v = fmaxf(v, 0.0f);
            atomicAdd(&pool_s[(size_t)batch[r] * N + col], v);
        }
    }
}

// ---------------- edge / pooling kernels ----------------

__global__ void k_count_edges(const int* __restrict__ ei, float* __restrict__ cnt, int E) {
    int e = blockIdx.x * blockDim.x + threadIdx.x;
    if (e < E) atomicAdd(&cnt[ei[E + e]], 1.0f);
}

// One thread per (edge, 4-float chunk): agg[dst] += xr[src]
__global__ void k_scatter_edges(const float* __restrict__ xr, const int* __restrict__ ei,
                                float* __restrict__ agg, int E, int F) {
    long t = (long)blockIdx.x * blockDim.x + threadIdx.x;
    int chunks = F >> 2;
    if (t >= (long)E * chunks) return;
    int e = (int)(t / chunks);
    int c = (int)(t % chunks) << 2;
    int s = ei[e];
    int d = ei[E + e];
    float4 v = *(const float4*)(xr + (size_t)s * F + c);
    float* o = agg + (size_t)d * F + c;
    atomicAdd(o + 0, v.x); atomicAdd(o + 1, v.y);
    atomicAdd(o + 2, v.z); atomicAdd(o + 3, v.w);
}

__global__ void k_pool_count(const int* __restrict__ batch, float* __restrict__ pool_n, int n) {
    int i = blockIdx.x * blockDim.x + threadIdx.x;
    if (i < n) atomicAdd(&pool_n[batch[i]], 1.0f);
}

__global__ void k_finish_pool(const float* __restrict__ pool_s, const float* __restrict__ pool_n,
                              float* __restrict__ pooled) {
    int idx = blockIdx.x * blockDim.x + threadIdx.x;
    if (idx >= N_GRAPHS * HID) return;
    int g = idx >> 9;
    pooled[idx] = pool_s[idx] / fmaxf(pool_n[g], 1.0f);
}

// hidden / cell heads: out[g,j] = pooled[g,:] . W[:,j] + b[j]   (tiny, VALU)
__global__ void k_dense_head(const float* __restrict__ pooled, const float* __restrict__ W,
                             const float* __restrict__ b, float* __restrict__ out) {
    int idx = blockIdx.x * blockDim.x + threadIdx.x;
    if (idx >= N_GRAPHS * HID) return;
    int g = idx >> 9, j = idx & 511;
    float acc = b[j];
    const float* p = pooled + (size_t)g * HID;
    for (int k = 0; k < HID; ++k) acc += p[k] * W[k * HID + j];
    out[idx] = acc;
}

// logits = log_softmax(hidden @ Wout + bout); one block (512 threads) per graph
__global__ void k_logits_lsm(const float* __restrict__ hidden, const float* __restrict__ Wout,
                             const float* __restrict__ bout, float* __restrict__ out) {
    __shared__ float red[VOCAB];
    __shared__ float red2[VOCAB];
    int g = blockIdx.x, v = threadIdx.x;
    float acc = bout[v];
    const float* h = hidden + (size_t)g * HID;
    for (int k = 0; k < HID; ++k) acc += h[k] * Wout[k * VOCAB + v];
    red[v] = acc;
    __syncthreads();
    for (int off = VOCAB / 2; off > 0; off >>= 1) {
        if (v < off) red[v] = fmaxf(red[v], red[v + off]);
        __syncthreads();
    }
    float m = red[0];
    __syncthreads();
    red2[v] = expf(acc - m);
    __syncthreads();
    for (int off = VOCAB / 2; off > 0; off >>= 1) {
        if (v < off) red2[v] += red2[v + off];
        __syncthreads();
    }
    float lse = m + logf(red2[0]);
    out[(size_t)g * VOCAB + v] = acc - lse;
}

// ---------------- host orchestration ----------------

static inline int cdiv(long a, int b) { return (int)((a + b - 1) / b); }

extern "C" void kernel_launch(void* const* d_in, const int* in_sizes, int n_in,
                              void* d_out, int out_size, void* d_ws, size_t ws_size,
                              hipStream_t stream) {
    const float* x       = (const float*)d_in[1];
    const int*   ei      = (const int*)d_in[2];
    const int*   batch   = (const int*)d_in[3];
    const float* W_root1 = (const float*)d_in[4];
    const float* W_rel1  = (const float*)d_in[5];
    const float* b1      = (const float*)d_in[6];
    const float* W_root2 = (const float*)d_in[7];
    const float* W_rel2  = (const float*)d_in[8];
    const float* b2      = (const float*)d_in[9];
    const float* Wh      = (const float*)d_in[10];
    const float* bh      = (const float*)d_in[11];
    const float* Wc      = (const float*)d_in[12];
    const float* bc      = (const float*)d_in[13];
    const float* Wout    = (const float*)d_in[14];
    const float* bout    = (const float*)d_in[15];

    char* ws = (char*)d_ws;
    size_t off = 0;
    auto take = [&](size_t bytes) { size_t r = off; off = (off + bytes + 255) & ~(size_t)255; return r; };
    float* cnt    = (float*)(ws + take((size_t)N_NODES * 4));              // in-degree (both layers)
    float* agg    = (float*)(ws + take((size_t)N_NODES * HID * 4));        // scatter dest (reused)
    float* Y      = (float*)(ws + take((size_t)N_NODES * HID * 4));        // rel-GEMM out (reused)
    bf16*  xb     = (bf16*) (ws + take((size_t)N_NODES * FEAT * 2));       // x in bf16
    bf16*  h1b    = (bf16*) (ws + take((size_t)N_NODES * EMB * 2));        // layer-1 output bf16
    bf16*  Bp     = (bf16*) (ws + take((size_t)EMB * HID * 2));            // packed weight (reused)
    float* pool_s = (float*)(ws + take((size_t)N_GRAPHS * HID * 4));
    float* pool_n = (float*)(ws + take((size_t)N_GRAPHS * 4));
    float* pooled = (float*)(ws + take((size_t)N_GRAPHS * HID * 4));

    const int T = 256;

    // degree (shared by both layers)
    k_zero_f32<<<cdiv(N_NODES, T), T, 0, stream>>>(cnt, N_NODES);
    k_count_edges<<<cdiv(N_EDGES, T), T, 0, stream>>>(ei, cnt, N_EDGES);

    // ---- layer 1 ----
    k_cvt_bf16<<<cdiv((long)N_NODES * FEAT, T), T, 0, stream>>>(x, xb, (long)N_NODES * FEAT);
    k_pack_b<<<cdiv(FEAT * EMB, T), T, 0, stream>>>(W_rel1, Bp, FEAT, EMB);
    {   // Y1rel = x @ W_rel1   [50000, 256]
        int tiles = (N_NODES / 16) * (EMB / 16);
        k_gemm_wmma<0><<<cdiv(tiles, 8), T, 0, stream>>>(xb, Bp, Y, nullptr, nullptr, nullptr,
                                                         nullptr, nullptr, nullptr,
                                                         N_NODES, EMB, FEAT);
    }
    k_zero_f32<<<cdiv((long)N_NODES * EMB, T), T, 0, stream>>>(agg, (long)N_NODES * EMB);
    k_scatter_edges<<<cdiv((long)N_EDGES * (EMB / 4), T), T, 0, stream>>>(Y, ei, agg, N_EDGES, EMB);
    k_pack_b<<<cdiv(FEAT * EMB, T), T, 0, stream>>>(W_root1, Bp, FEAT, EMB);
    {   // h1 = relu(x @ W_root1 + agg/cnt + b1) -> bf16
        int tiles = (N_NODES / 16) * (EMB / 16);
        k_gemm_wmma<1><<<cdiv(tiles, 8), T, 0, stream>>>(xb, Bp, nullptr, agg, cnt, b1,
                                                         h1b, nullptr, nullptr,
                                                         N_NODES, EMB, FEAT);
    }

    // ---- layer 2 ----
    k_pack_b<<<cdiv(EMB * HID, T), T, 0, stream>>>(W_rel2, Bp, EMB, HID);
    {   // Y2rel = h1 @ W_rel2  [50000, 512]
        int tiles = (N_NODES / 16) * (HID / 16);
        k_gemm_wmma<0><<<cdiv(tiles, 8), T, 0, stream>>>(h1b, Bp, Y, nullptr, nullptr, nullptr,
                                                         nullptr, nullptr, nullptr,
                                                         N_NODES, HID, EMB);
    }
    k_zero_f32<<<cdiv((long)N_NODES * HID, T), T, 0, stream>>>(agg, (long)N_NODES * HID);
    k_scatter_edges<<<cdiv((long)N_EDGES * (HID / 4), T), T, 0, stream>>>(Y, ei, agg, N_EDGES, HID);

    // pooling accumulators
    k_zero_f32<<<cdiv(N_GRAPHS * HID, T), T, 0, stream>>>(pool_s, N_GRAPHS * HID);
    k_zero_f32<<<1, T, 0, stream>>>(pool_n, N_GRAPHS);
    k_pool_count<<<cdiv(N_NODES, T), T, 0, stream>>>(batch, pool_n, N_NODES);

    k_pack_b<<<cdiv(EMB * HID, T), T, 0, stream>>>(W_root2, Bp, EMB, HID);
    {   // h2 = relu(h1 @ W_root2 + agg/cnt + b2), fused atomic pool-sum
        int tiles = (N_NODES / 16) * (HID / 16);
        k_gemm_wmma<2><<<cdiv(tiles, 8), T, 0, stream>>>(h1b, Bp, nullptr, agg, cnt, b2,
                                                         nullptr, batch, pool_s,
                                                         N_NODES, HID, EMB);
    }
    k_finish_pool<<<cdiv(N_GRAPHS * HID, T), T, 0, stream>>>(pool_s, pool_n, pooled);

    // ---- heads: d_out = [logits | hidden | cell] ----
    float* out_logits = (float*)d_out;
    float* out_hidden = out_logits + (size_t)N_GRAPHS * VOCAB;
    float* out_cell   = out_hidden + (size_t)N_GRAPHS * HID;
    k_dense_head<<<cdiv(N_GRAPHS * HID, T), T, 0, stream>>>(pooled, Wh, bh, out_hidden);
    k_dense_head<<<cdiv(N_GRAPHS * HID, T), T, 0, stream>>>(pooled, Wc, bc, out_cell);
    k_logits_lsm<<<N_GRAPHS, VOCAB, 0, stream>>>(out_hidden, Wout, bout, out_logits);
}
